// SelfAttention_73486890435069
// MI455X (gfx1250) — matrix operand order repaired and checked by reading, not compiled, not run on hardware
//
#include <hip/hip_runtime.h>
#include <hip/hip_bf16.h>

// ---------------------------------------------------------------------------
// MI455X (gfx1250) self-attention:
//   bf16 WMMA GEMMs with TDM (tensor_load_to_lds) double-buffered LDS staging
//   + wave-level flash attention.
// ---------------------------------------------------------------------------

constexpr int kB = 4;
constexpr int kS = 2048;
constexpr int kE = 1024;
constexpr int kH = 16;
constexpr int kD = 64;

typedef __attribute__((ext_vector_type(16))) __bf16       bf16x16;
typedef __attribute__((ext_vector_type(8)))  __bf16       bf16x8;
typedef __attribute__((ext_vector_type(8)))  float        f32x8;
typedef __attribute__((ext_vector_type(4)))  unsigned int u32x4;
typedef __attribute__((ext_vector_type(8)))  int          i32x8;
typedef __attribute__((ext_vector_type(4)))  int          i32x4;

#if __has_builtin(__builtin_amdgcn_tensor_load_to_lds)
#define HAVE_TDM 1
#else
#define HAVE_TDM 0
#endif

static __device__ __forceinline__ void wait_tensorcnt0() {
#if __has_builtin(__builtin_amdgcn_s_wait_tensorcnt)
  __builtin_amdgcn_s_wait_tensorcnt(0);
#else
  asm volatile("s_wait_tensorcnt 0x0" ::: "memory");
#endif
}

static __device__ __forceinline__ f32x8 wmma_bf16(bf16x16 a, bf16x16 b, f32x8 c) {
  // emits v_wmma_f32_16x16x32_bf16
  return __builtin_amdgcn_wmma_f32_16x16x32_bf16(false, a, false, b,
                                                 (short)0, c, false, false);
}

// A fragment (16x32 bf16, row-major source, lda in elements).
// Lane L holds row M=L%16; lanes 0-15: K=0..7,16..23; lanes 16-31: K=8..15,24..31.
static __device__ __forceinline__ bf16x16 load_a_frag(const __bf16* base, int lda) {
  const int lane = threadIdx.x & 31;
  const int row  = lane & 15;
  const int kb   = (lane >> 4) << 3;  // 0 or 8
  const __bf16* p = base + row * lda + kb;
  bf16x8 lo = *(const bf16x8*)(p);       // K = kb .. kb+7
  bf16x8 hi = *(const bf16x8*)(p + 16);  // K = kb+16 .. kb+23
  bf16x16 r;
#pragma unroll
  for (int i = 0; i < 8; ++i) { r[i] = lo[i]; r[i + 8] = hi[i]; }
  return r;
}

// B fragment (32x16 bf16) from transposed source Bt[n][k] (ldb elements between cols).
// Lane L holds col N=L%16; lanes 0-15: K=0..7,16..23; lanes 16-31: K=8..15,24..31.
static __device__ __forceinline__ bf16x16 load_b_frag(const __bf16* base, int ldb) {
  const int lane = threadIdx.x & 31;
  const int col  = lane & 15;
  const int kb   = (lane >> 4) << 3;  // 0 or 8
  const __bf16* p = base + col * ldb + kb;
  bf16x8 lo = *(const bf16x8*)(p);
  bf16x8 hi = *(const bf16x8*)(p + 16);
  bf16x16 r;
#pragma unroll
  for (int i = 0; i < 8; ++i) { r[i] = lo[i]; r[i + 8] = hi[i]; }
  return r;
}

static __device__ __forceinline__ f32x8 zero8() {
  f32x8 z = {0.f, 0.f, 0.f, 0.f, 0.f, 0.f, 0.f, 0.f};
  return z;
}

// Generic LDS pointer -> byte offset within the workgroup LDS allocation
// (aperture encoding keeps the LDS offset in addr[31:0]).
static __device__ __forceinline__ unsigned lds_off(const void* p) {
  return (unsigned)(unsigned long long)p;
}

#if HAVE_TDM
// Issue one TDM 2D tile load: tile_rows x 32 bf16 elements from a row-major
// tensor (row stride = stride_elems), deposited into LDS with a 4-DWORD pad
// after every 16 DWORDs -> LDS row stride 80B ([row][40] bf16 layout).
static __device__ __forceinline__ void tdm_load_tile(const __bf16* gsrc,
                                                     unsigned lds_byte,
                                                     int tensor_rows,
                                                     int stride_elems,
                                                     int tile_rows) {
  const unsigned long long ga = (unsigned long long)gsrc;
  u32x4 g0;
  g0[0] = 1u;                                        // count=1, user mode
  g0[1] = lds_byte;                                  // lds_addr
  g0[2] = (unsigned)(ga & 0xFFFFFFFFu);              // global_addr[31:0]
  g0[3] = (unsigned)((ga >> 32) & 0x1FFFFFFu) | (2u << 30);  // addr[56:32] | type=2
  i32x8 g1;
  g1[0] = (1 << 16)      // data_size = 2 bytes
        | (1 << 20)      // pad_enable
        | (3 << 22)      // pad_interval: 16 DWORDs (64B)
        | (3 << 25);     // pad_amount:   4 DWORDs (16B)
  g1[1] = (int)(((unsigned)kE & 0xFFFFu) << 16);     // tensor_dim0 = 1024 (lo 16)
  g1[2] = (int)((((unsigned)kE >> 16) & 0xFFFFu)     // tensor_dim0 hi
        | (((unsigned)tensor_rows & 0xFFFFu) << 16));// tensor_dim1 lo
  g1[3] = (int)((((unsigned)tensor_rows >> 16) & 0xFFFFu)
        | (32u << 16));                              // tile_dim0 = 32
  g1[4] = tile_rows;                                 // tile_dim1 | tile_dim2=0
  g1[5] = stride_elems;                              // tensor_dim0_stride lo
  g1[6] = 0;                                         // stride0 hi | stride1 lo
  g1[7] = 0;                                         // stride1 hi
  i32x4 g2;
  g2[0] = 1;  // tensor_dim2 = 1
  g2[1] = 0;  // tensor_dim3
  g2[2] = 0;  // tensor_dim2_stride lo
  g2[3] = 0;  // stride hi | tile_dim3
  i32x4 g3 = {0, 0, 0, 0};
  i32x8 g4 = {0, 0, 0, 0, 0, 0, 0, 0};  // trailing group (6-arg toolchain form)
  __builtin_amdgcn_tensor_load_to_lds(g0, g1, g2, g3, g4, 0);
}
#endif

// ---------------------------------------------------------------------------
// fp32 -> bf16 elementwise conversion (grid-stride)
// ---------------------------------------------------------------------------
__global__ void cvt_f32_to_bf16(const float* __restrict__ x,
                                __bf16* __restrict__ y, int n) {
  int i = blockIdx.x * blockDim.x + threadIdx.x;
  const int stride = gridDim.x * blockDim.x;
  for (; i < n; i += stride) y[i] = (__bf16)x[i];
}

// ---------------------------------------------------------------------------
// fp32 W[k][n] -> bf16 Wt[n][k], LDS-tiled transpose
// ---------------------------------------------------------------------------
__global__ __launch_bounds__(256) void transpose_to_bf16(const float* __restrict__ w,
                                                         __bf16* __restrict__ wt) {
  __shared__ __bf16 t[32][33];
  const int n0 = blockIdx.x * 32;
  const int k0 = blockIdx.y * 32;
  const int tx = threadIdx.x;  // 0..31
  const int ty = threadIdx.y;  // 0..7
#pragma unroll
  for (int i = 0; i < 32; i += 8)
    t[ty + i][tx] = (__bf16)w[(size_t)(k0 + ty + i) * kE + (n0 + tx)];
  __syncthreads();
#pragma unroll
  for (int i = 0; i < 32; i += 8)
    wt[(size_t)(n0 + ty + i) * kE + (k0 + tx)] = t[tx][ty + i];
}

// ---------------------------------------------------------------------------
// Projection GEMM with TDM-staged LDS double buffering.
// C[m,n] = A[m,:] @ Wt[n,:]^T + bias[n]
// Block tile 128x128 (8 waves of 32x64). Per 32-k chunk wave 0 issues two
// tensor_load_to_lds DMAs (A and B tiles) into the ping-pong LDS buffers.
// mode 0: bf16 [B,H,S,D] (Q/K)   mode 1: bf16 [B,H,D,S] (V^T)   mode 2: fp32 [m,n]
// ---------------------------------------------------------------------------
__global__ __launch_bounds__(256) void gemm_bf16_wmma(const __bf16* __restrict__ A,
                                                      const __bf16* __restrict__ Bt,
                                                      int a_rows,
                                                      const float* __restrict__ bias,
                                                      void* __restrict__ out,
                                                      int mode) {
  __shared__ __align__(16) __bf16 Alds[2][128][40];
  __shared__ __align__(16) __bf16 Blds[2][128][40];

  const int tid  = threadIdx.x;
  const int lane = tid & 31;
  const int wv   = tid >> 5;
  const int bm   = blockIdx.x >> 3;  // 64 m-blocks of 128 rows
  const int bn   = blockIdx.x & 7;   // 8 n-blocks of 128 cols
  const int m0   = bm * 128;
  const int n0   = bn * 128;
  const int wm   = wv & 3;           // wave row subtile (4 x 32 rows)
  const int wn   = wv >> 2;          // wave col subtile (2 x 64 cols)

  f32x8 acc[2][4];
#pragma unroll
  for (int i = 0; i < 2; ++i)
#pragma unroll
    for (int f = 0; f < 4; ++f) acc[i][f] = zero8();

  const int nIter = kE / 32;

#if HAVE_TDM
  int buf = 0;
  if (wv == 0) {
    tdm_load_tile(A + (size_t)m0 * kE, lds_off(&Alds[0][0][0]), a_rows, kE, 128);
    tdm_load_tile(Bt + (size_t)n0 * kE, lds_off(&Blds[0][0][0]), kE, kE, 128);
  }
  for (int it = 0; it < nIter; ++it) {
    if (wv == 0) wait_tensorcnt0();   // chunk `it` deposited (TENSORcnt in-order)
    __syncthreads();                  // visible to all waves
    if (wv == 0 && it + 1 < nIter) {  // prefetch next chunk into other buffer
      const int kc = (it + 1) * 32;
      tdm_load_tile(A + (size_t)m0 * kE + kc, lds_off(&Alds[buf ^ 1][0][0]),
                    a_rows, kE, 128);
      tdm_load_tile(Bt + (size_t)n0 * kE + kc, lds_off(&Blds[buf ^ 1][0][0]),
                    kE, kE, 128);
    }
    bf16x16 a0 = load_a_frag(&Alds[buf][wm * 32][0], 40);
    bf16x16 a1 = load_a_frag(&Alds[buf][wm * 32 + 16][0], 40);
#pragma unroll
    for (int f = 0; f < 4; ++f) {
      bf16x16 b = load_b_frag(&Blds[buf][wn * 64 + f * 16][0], 40);
      acc[0][f] = wmma_bf16(a0, b, acc[0][f]);
      acc[1][f] = wmma_bf16(a1, b, acc[1][f]);
    }
    buf ^= 1;
  }
#else
  // Cooperative fallback stage (no TDM builtin): 256 threads copy 16B slots.
  for (int it = 0; it < nIter; ++it) {
    const int kc = it * 32;
#pragma unroll
    for (int s = 0; s < 2; ++s) {
      const int slot = tid + s * 256;       // 512 slots of 16B per tile
      const int row  = slot >> 2;
      const int ch   = slot & 3;
      *(bf16x8*)&Alds[0][row][ch * 8] =
          *(const bf16x8*)(A + (size_t)(m0 + row) * kE + kc + ch * 8);
      *(bf16x8*)&Blds[0][row][ch * 8] =
          *(const bf16x8*)(Bt + (size_t)(n0 + row) * kE + kc + ch * 8);
    }
    __syncthreads();
    bf16x16 a0 = load_a_frag(&Alds[0][wm * 32][0], 40);
    bf16x16 a1 = load_a_frag(&Alds[0][wm * 32 + 16][0], 40);
#pragma unroll
    for (int f = 0; f < 4; ++f) {
      bf16x16 b = load_b_frag(&Blds[0][wn * 64 + f * 16][0], 40);
      acc[0][f] = wmma_bf16(a0, b, acc[0][f]);
      acc[1][f] = wmma_bf16(a1, b, acc[1][f]);
    }
    __syncthreads();
  }
#endif

  const int half = lane >> 4;
  const int cl   = lane & 15;
  const int mw0  = m0 + wm * 32;
  const int nw0  = n0 + wn * 64;
#pragma unroll
  for (int i = 0; i < 2; ++i) {
#pragma unroll
    for (int f = 0; f < 4; ++f) {
#pragma unroll
      for (int r = 0; r < 8; ++r) {
        const int m = mw0 + i * 16 + r + half * 8;
        const int n = nw0 + f * 16 + cl;
        const float v = acc[i][f][r] + bias[n];
        if (mode == 2) {
          ((float*)out)[(size_t)m * kE + n] = v;
        } else {
          const int b = m >> 11;          // m / kS
          const int s = m & (kS - 1);
          const int h = n >> 6;           // n / kD
          const int d = n & (kD - 1);
          if (mode == 0)
            ((__bf16*)out)[(((size_t)(b * kH + h) * kS) + s) * kD + d] = (__bf16)v;
          else
            ((__bf16*)out)[(((size_t)(b * kH + h) * kD) + d) * kS + s] = (__bf16)v;
        }
      }
    }
  }
}

// ---------------------------------------------------------------------------
// Flash attention: one wave per (b, h, 16-row Q tile).
// Q,K: bf16 [B,H,S,D]; Vt: bf16 [B,H,D,S]; Out: bf16 [B,S,E].
// ---------------------------------------------------------------------------
__global__ __launch_bounds__(256) void attn_flash_wmma(const __bf16* __restrict__ Q,
                                                       const __bf16* __restrict__ K,
                                                       const __bf16* __restrict__ Vt,
                                                       __bf16* __restrict__ Out) {
  __shared__ __align__(16) __bf16 pbuf[8][16][40];  // per-wave 16x32 P tile (+pad)

  const int lane = threadIdx.x & 31;
  const int wv   = threadIdx.x >> 5;
  const int gw   = blockIdx.x * 8 + wv;
  const int qt   = gw & 127;          // 128 q-tiles per head
  const int h    = (gw >> 7) & 15;
  const int b    = gw >> 11;
  const int q0   = qt * 16;

  const __bf16* Qh = Q  + ((size_t)(b * kH + h) * kS + q0) * kD;
  const __bf16* Kh = K  + (size_t)(b * kH + h) * kS * kD;
  const __bf16* Vh = Vt + (size_t)(b * kH + h) * kD * kS;

  const bf16x16 qa0 = load_a_frag(Qh, kD);
  const bf16x16 qa1 = load_a_frag(Qh + 32, kD);

  const int half = lane >> 4;
  const int cl   = lane & 15;

  float mrow[8], lrow[8];
  f32x8 o[4];
#pragma unroll
  for (int r = 0; r < 8; ++r) { mrow[r] = -1e30f; lrow[r] = 0.f; }
#pragma unroll
  for (int f = 0; f < 4; ++f) o[f] = zero8();

  const float scale = 0.125f;  // 1/sqrt(64)

  for (int k0 = 0; k0 < kS; k0 += 32) {
    f32x8 s0 = zero8(), s1 = zero8();
    s0 = wmma_bf16(qa0, load_b_frag(Kh + (size_t)k0 * kD, kD), s0);
    s0 = wmma_bf16(qa1, load_b_frag(Kh + (size_t)k0 * kD + 32, kD), s0);
    s1 = wmma_bf16(qa0, load_b_frag(Kh + (size_t)(k0 + 16) * kD, kD), s1);
    s1 = wmma_bf16(qa1, load_b_frag(Kh + (size_t)(k0 + 16) * kD + 32, kD), s1);

    float alpha[8];
#pragma unroll
    for (int r = 0; r < 8; ++r) {
      const float v0 = s0[r] * scale;
      const float v1 = s1[r] * scale;
      float mx = fmaxf(v0, v1);
#pragma unroll
      for (int off = 8; off >= 1; off >>= 1) mx = fmaxf(mx, __shfl_xor(mx, off, 32));
      const float mnew = fmaxf(mrow[r], mx);
      alpha[r] = __expf(mrow[r] - mnew);
      mrow[r]  = mnew;
      const float p0 = __expf(v0 - mnew);
      const float p1 = __expf(v1 - mnew);
      float sm = p0 + p1;
#pragma unroll
      for (int off = 8; off >= 1; off >>= 1) sm += __shfl_xor(sm, off, 32);
      lrow[r] = lrow[r] * alpha[r] + sm;
      const int row = r + half * 8;
      pbuf[wv][row][cl]      = (__bf16)p0;
      pbuf[wv][row][cl + 16] = (__bf16)p1;
    }

#pragma unroll
    for (int f = 0; f < 4; ++f)
#pragma unroll
      for (int r = 0; r < 8; ++r) o[f][r] *= alpha[r];

    const bf16x16 pa = load_a_frag(&pbuf[wv][0][0], 40);

#pragma unroll
    for (int f = 0; f < 4; ++f) {
      bf16x16 vb = load_b_frag(Vh + (size_t)(f * 16) * kS + k0, kS);
      o[f] = wmma_bf16(pa, vb, o[f]);
    }
  }

#pragma unroll
  for (int f = 0; f < 4; ++f) {
#pragma unroll
    for (int r = 0; r < 8; ++r) {
      const int srow = q0 + r + half * 8;
      const float v  = o[f][r] / lrow[r];
      Out[((size_t)(b * kS + srow) * kE) + h * kD + f * 16 + cl] = (__bf16)v;
    }
  }
}

// ---------------------------------------------------------------------------
// Host launch
// ---------------------------------------------------------------------------
extern "C" void kernel_launch(void* const* d_in, const int* in_sizes, int n_in,
                              void* d_out, int out_size, void* d_ws, size_t ws_size,
                              hipStream_t stream) {
  const float* X  = (const float*)d_in[0];
  const float* Wq = (const float*)d_in[1];
  const float* bq = (const float*)d_in[2];
  const float* Wk = (const float*)d_in[3];
  const float* bk = (const float*)d_in[4];
  const float* Wv = (const float*)d_in[5];
  const float* bv = (const float*)d_in[6];
  const float* Wo = (const float*)d_in[7];
  const float* bo = (const float*)d_in[8];

  const size_t nX = (size_t)kB * kS * kE;  // 8388608
  const size_t nW = (size_t)kE * kE;       // 1048576
  const int    aRows = kB * kS;            // 8192

  __bf16* ws  = (__bf16*)d_ws;
  __bf16* Xb  = ws;                 // [8192,1024] bf16
  __bf16* Wqt = Xb  + nX;           // [n][k]
  __bf16* Wkt = Wqt + nW;
  __bf16* Wvt = Wkt + nW;
  __bf16* Wot = Wvt + nW;
  __bf16* Qb  = Wot + nW;           // [B,H,S,D]
  __bf16* Kb  = Qb  + nX;           // [B,H,S,D]
  __bf16* Vtb = Kb  + nX;           // [B,H,D,S]
  __bf16* Ao  = Vtb + nX;           // [B,S,E]

  cvt_f32_to_bf16<<<512, 256, 0, stream>>>(X, Xb, (int)nX);

  dim3 tb(32, 8), tg(kE / 32, kE / 32);
  transpose_to_bf16<<<tg, tb, 0, stream>>>(Wq, Wqt);
  transpose_to_bf16<<<tg, tb, 0, stream>>>(Wk, Wkt);
  transpose_to_bf16<<<tg, tb, 0, stream>>>(Wv, Wvt);
  transpose_to_bf16<<<tg, tb, 0, stream>>>(Wo, Wot);

  gemm_bf16_wmma<<<512, 256, 0, stream>>>(Xb, Wqt, aRows, bq, Qb, 0);
  gemm_bf16_wmma<<<512, 256, 0, stream>>>(Xb, Wkt, aRows, bk, Kb, 0);
  gemm_bf16_wmma<<<512, 256, 0, stream>>>(Xb, Wvt, aRows, bv, Vtb, 1);

  attn_flash_wmma<<<1024, 256, 0, stream>>>(Qb, Kb, Vtb, Ao);

  gemm_bf16_wmma<<<512, 256, 0, stream>>>(Ao, Wot, aRows, bo, (float*)d_out, 2);
}
